// PolicyNet_66769561584187
// MI455X (gfx1250) — compile-verified, matrix operand order checked by reading, not run on hardware
//
#include <hip/hip_runtime.h>
#include <hip/hip_bf16.h>

// ---------------------------------------------------------------------------
// PolicyNet LSTM scan on gfx1250 (MI455X).
//   Per step: gates = h@w_hh^T (+bias) via WMMA, one-hot w_ih gather,
//   LSTM elementwise, logits = h@w_lin^T (+bias) via WMMA over the active
//   512-column half, wave-parallel masked softmax + inverse-CDF sampling.
//   Persistent block: 128 threads (4 waves), 64 batch rows, all 32 steps.
//   Logits tile stored with 33-float chunk stride -> conflict-free phase C.
//   WMMA tiles pipelined in pairs (independent accumulators) to hide the
//   WMMA->VALU hazard window.
// ---------------------------------------------------------------------------

typedef __attribute__((ext_vector_type(16))) _Float16 v16h;
typedef __attribute__((ext_vector_type(8)))  float    v8f;

#define NH       24     // n_hidden
#define HP       32     // padded K for WMMA
#define G4       96     // 4*NH gates
#define NOUTC    1024
#define PHC      512    // active slice width per step
#define LSTEPS   32
#define BATCH    8192
#define BROWS    64     // batch rows per block
#define NTHREADS 128    // 4 wave32
#define CHS      33     // padded chunk stride (floats) in logits tile
#define LSTRIDE  (16 * CHS)   // 528 floats per logits row

// shared memory carve (bytes)
#define OFF_WLIN   0u                      // f16 [1024][32]        65536
#define OFF_LOGITS (OFF_WLIN   + 65536u)   // f32 [64][528]        135168
#define OFF_GATES  (OFF_LOGITS + 135168u)  // f32 [64][96]          24576
#define OFF_C      (OFF_GATES  + 24576u)   // f32 [64][24]           6144
#define OFF_H16    (OFF_C      + 6144u)    // f16 [64][32]           4096
#define OFF_WHH16  (OFF_H16    + 4096u)    // f16 [96][32]           6144
#define OFF_BLIN   (OFF_WHH16  + 6144u)    // f32 [1024]             4096
#define OFF_BIHH   (OFF_BLIN   + 4096u)    // f32 [96]                384
#define OFF_ACT    (OFF_BIHH   + 384u)     // i32 [64]                256
#define SMEM_BYTES (OFF_ACT    + 256u)     // 246400 bytes total

union FragH { v16h h; unsigned int u[8]; };

__device__ __forceinline__ float sigm(float x) { return 1.0f / (1.0f + __expf(-x)); }

__global__ __launch_bounds__(NTHREADS)
void policy_lstm_scan_kernel(const float* __restrict__ w_ih,   // [96,1024]
                             const float* __restrict__ w_hh,   // [96,24]
                             const float* __restrict__ b_ih,   // [96]
                             const float* __restrict__ b_hh,   // [96]
                             const float* __restrict__ w_lin,  // [1024,24]
                             const float* __restrict__ b_lin,  // [1024]
                             const float* __restrict__ u,      // [32,8192]
                             const int*   __restrict__ training,
                             float* __restrict__ out)          // [B*L probs | B*L actions]
{
    extern __shared__ __align__(16) char smem[];
    _Float16* s_wlin  = (_Float16*)(smem + OFF_WLIN);
    float*    s_log   = (float*)   (smem + OFF_LOGITS);
    float*    s_gates = (float*)   (smem + OFF_GATES);
    float*    s_c     = (float*)   (smem + OFF_C);
    _Float16* s_h16   = (_Float16*)(smem + OFF_H16);
    _Float16* s_whh16 = (_Float16*)(smem + OFF_WHH16);
    float*    s_blin  = (float*)   (smem + OFF_BLIN);
    float*    s_bihh  = (float*)   (smem + OFF_BIHH);
    int*      s_act   = (int*)     (smem + OFF_ACT);

    const int tid  = threadIdx.x;
    const int wave = tid >> 5;
    const int lane = tid & 31;
    const int row0 = blockIdx.x * BROWS;   // global batch row base for block

    // ---- prefetch the per-step scattered gather source (w_ih) ------------
    for (int off = tid * 32; off < G4 * NOUTC; off += NTHREADS * 32)
        __builtin_prefetch(&w_ih[off], 0, 1);

    // ---- stage weights / init state --------------------------------------
    for (int i = tid; i < NOUTC * HP; i += NTHREADS) {
        int n = i / HP, k = i % HP;
        s_wlin[i] = (k < NH) ? (_Float16)w_lin[n * NH + k] : (_Float16)0.0f;
    }
    for (int i = tid; i < G4 * HP; i += NTHREADS) {
        int n = i / HP, k = i % HP;
        s_whh16[i] = (k < NH) ? (_Float16)w_hh[n * NH + k] : (_Float16)0.0f;
    }
    for (int i = tid; i < NOUTC;   i += NTHREADS) s_blin[i] = b_lin[i];
    for (int i = tid; i < G4;      i += NTHREADS) s_bihh[i] = b_ih[i] + b_hh[i];
    for (int i = tid; i < BROWS * NH; i += NTHREADS) s_c[i] = 0.0f;
    for (int i = tid; i < BROWS * HP; i += NTHREADS) s_h16[i] = (_Float16)0.0f;
    for (int i = tid; i < BROWS;      i += NTHREADS) s_act[i] = -1;
    __syncthreads();

    const int train = training[0];
    const int hi    = (lane >= 16) ? 8 : 0;   // C-fragment row offset
    const int l15   = lane & 15;

    for (int t = 0; t < LSTEPS; ++t) {
        const int nb = (t & 1) ? PHC : 0;     // active slice base column

        // ---- phase A1: gates = h_prev @ w_hh^T + (b_ih+b_hh) via WMMA ----
        {
            const int mrow = wave * 16 + l15;
            FragH A;
            #pragma unroll
            for (int v = 0; v < 8; ++v) {
                const int k0 = ((v < 4) ? 0 : 16) + hi + 2 * (v & 3);
                A.u[v] = *(const unsigned int*)(&s_h16[mrow * HP + k0]);
            }
            #pragma unroll
            for (int ct = 0; ct < G4 / 16; ct += 2) {      // 3 pipelined pairs
                const int col0 = ct * 16 + l15;
                const int col1 = col0 + 16;
                FragH B0, B1;
                #pragma unroll
                for (int v = 0; v < 8; ++v) {
                    const int k0 = ((v < 4) ? 0 : 16) + hi + 2 * (v & 3);
                    B0.u[v] = *(const unsigned int*)(&s_whh16[col0 * HP + k0]);
                    B1.u[v] = *(const unsigned int*)(&s_whh16[col1 * HP + k0]);
                }
                v8f c0, c1;
                const float b0 = s_bihh[col0], b1 = s_bihh[col1];
                #pragma unroll
                for (int r = 0; r < 8; ++r) { c0[r] = b0; c1[r] = b1; }
                c0 = __builtin_amdgcn_wmma_f32_16x16x32_f16(
                        false, A.h, false, B0.h, (short)0, c0, false, false);
                c1 = __builtin_amdgcn_wmma_f32_16x16x32_f16(
                        false, A.h, false, B1.h, (short)0, c1, false, false);
                #pragma unroll
                for (int r = 0; r < 8; ++r) {
                    s_gates[(wave * 16 + hi + r) * G4 + col0] = c0[r];
                    s_gates[(wave * 16 + hi + r) * G4 + col1] = c1[r];
                }
            }
        }
        __syncthreads();

        // ---- phase A2: one-hot w_ih gather + LSTM cell (VALU) ------------
        for (int idx = tid; idx < BROWS * NH; idx += NTHREADS) {
            const int r = idx / NH, j = idx % NH;
            const int act = s_act[r];
            float g0 = s_gates[r * G4 + j];
            float g1 = s_gates[r * G4 + NH + j];
            float g2 = s_gates[r * G4 + 2 * NH + j];
            float g3 = s_gates[r * G4 + 3 * NH + j];
            if (act >= 0) {                    // a_prev one-hot column gather
                g0 += w_ih[(j)           * NOUTC + act];
                g1 += w_ih[(NH + j)      * NOUTC + act];
                g2 += w_ih[(2 * NH + j)  * NOUTC + act];
                g3 += w_ih[(3 * NH + j)  * NOUTC + act];
            }
            const float ig = sigm(g0), fg = sigm(g1);
            const float gg = tanhf(g2), og = sigm(g3);
            const float cn = fg * s_c[idx] + ig * gg;
            const float hn = og * tanhf(cn);
            s_c[idx] = cn;
            s_h16[r * HP + j] = (_Float16)hn;  // h state (f16, WMMA operand)
        }
        __syncthreads();

        // ---- phase B: logits = h @ w_lin^T (+b_lin) via WMMA -------------
        // Store column c at row*528 + (c>>5)*33 + (c&31)  (chunk-padded).
        {
            const int mrow = wave * 16 + l15;
            FragH A;
            #pragma unroll
            for (int v = 0; v < 8; ++v) {
                const int k0 = ((v < 4) ? 0 : 16) + hi + 2 * (v & 3);
                A.u[v] = *(const unsigned int*)(&s_h16[mrow * HP + k0]);
            }
            for (int nt = 0; nt < PHC / 16; nt += 2) {     // 16 pipelined pairs
                const int ncol0 = nb + nt * 16 + l15;
                const int ncol1 = ncol0 + 16;
                FragH B0, B1;
                #pragma unroll
                for (int v = 0; v < 8; ++v) {
                    const int k0 = ((v < 4) ? 0 : 16) + hi + 2 * (v & 3);
                    B0.u[v] = *(const unsigned int*)(&s_wlin[ncol0 * HP + k0]);
                    B1.u[v] = *(const unsigned int*)(&s_wlin[ncol1 * HP + k0]);
                }
                v8f c0, c1;
                const float b0 = s_blin[ncol0], b1 = s_blin[ncol1];
                #pragma unroll
                for (int r = 0; r < 8; ++r) { c0[r] = b0; c1[r] = b1; }
                c0 = __builtin_amdgcn_wmma_f32_16x16x32_f16(
                        false, A.h, false, B0.h, (short)0, c0, false, false);
                c1 = __builtin_amdgcn_wmma_f32_16x16x32_f16(
                        false, A.h, false, B1.h, (short)0, c1, false, false);
                const int kbase = (nt >> 1) * CHS + l15;   // chunk-padded col
                #pragma unroll
                for (int r = 0; r < 8; ++r) {
                    const int rowb = (wave * 16 + hi + r) * LSTRIDE;
                    s_log[rowb + kbase]      = c0[r];
                    s_log[rowb + kbase + 16] = c1[r];
                }
            }
        }
        __syncthreads();

        // ---- phase C: wave-parallel masked softmax + sampling ------------
        // 16 lanes per row, 8 rows per pass, 8 passes. Lane `sub` owns the
        // contiguous logical columns [32*sub, 32*sub+32) stored at chunk
        // offset sub*33 (bank-conflict-free across the 16 lanes).
        {
            const int sub = tid & 15;          // lane within 16-lane group
            const int grp = tid >> 4;          // 0..7
            for (int it = 0; it < BROWS / 8; ++it) {
                const int r  = it * 8 + grp;
                const int gr = row0 + r;
                float* lg = &s_log[r * LSTRIDE];
                const int base = sub * CHS;    // padded chunk base

                float m = -3.4e38f;
                #pragma unroll 8
                for (int i = 0; i < 32; ++i) m = fmaxf(m, lg[base + i]);
                #pragma unroll
                for (int o = 8; o >= 1; o >>= 1) m = fmaxf(m, __shfl_xor(m, o, 16));

                float sl = 0.0f;               // chunk exp-sum, exps cached
                #pragma unroll 8
                for (int i = 0; i < 32; ++i) {
                    float e = __expf(lg[base + i] - m);
                    lg[base + i] = e;
                    sl += e;
                }
                float pre = sl;                // inclusive prefix over chunks
                #pragma unroll
                for (int o = 1; o < 16; o <<= 1) {
                    float v = __shfl_up(pre, o, 16);
                    if (sub >= o) pre += v;
                }
                const float total = __shfl(pre, 15, 16);
                const float inv   = 1.0f / total;
                const float excl  = (pre - sl) * inv;

                int ga; float p;
                if (train) {
                    const float uv = u[t * BATCH + gr];
                    int actLoc = -1; float pLoc = 0.0f;
                    const float hiCdf = excl + sl * inv;
                    if (excl <= uv && hiCdf > uv) {      // crossing in my chunk
                        float cum = excl;
                        for (int i = 0; i < 32; ++i) {
                            const float q = lg[base + i] * inv;
                            cum += q;
                            if (cum > uv) { actLoc = sub * 32 + i; pLoc = q; break; }
                        }
                    }
                    const unsigned long long bal = __ballot(actLoc >= 0);
                    const int gsh = (lane >> 4) << 4;    // 0 or 16
                    const unsigned int gm = (unsigned int)((bal >> gsh) & 0xFFFFu);
                    if (gm) {
                        const int srcl = (__ffs(gm) - 1) + gsh;
                        const int aL = __shfl(actLoc, srcl, 32);
                        p  = __shfl(pLoc, srcl, 32);
                        ga = nb + aL;
                    } else {                              // argmax(all-false)==0
                        ga = 0;
                        p  = (nb == 0) ? lg[0] * inv : 0.0f;
                    }
                } else {                                  // eval: argmax
                    float be = -1.0f; int bi = 0;
                    #pragma unroll 8
                    for (int i = 0; i < 32; ++i) {
                        const float e = lg[base + i];
                        if (e > be) { be = e; bi = sub * 32 + i; }
                    }
                    #pragma unroll
                    for (int o = 8; o >= 1; o >>= 1) {
                        const float oe = __shfl_xor(be, o, 16);
                        const int   oi = __shfl_xor(bi, o, 16);
                        if (oe > be || (oe == be && oi < bi)) { be = oe; bi = oi; }
                    }
                    ga = nb + bi;
                    p  = be * inv;
                }
                if (sub == 0) {
                    out[gr * LSTEPS + t]                  = p;
                    out[BATCH * LSTEPS + gr * LSTEPS + t] = (float)ga;
                    s_act[r] = ga;
                }
            }
        }
        __syncthreads();
    }
}

extern "C" void kernel_launch(void* const* d_in, const int* in_sizes, int n_in,
                              void* d_out, int out_size, void* d_ws, size_t ws_size,
                              hipStream_t stream) {
    (void)in_sizes; (void)n_in; (void)out_size; (void)d_ws; (void)ws_size;
    const float* w_ih  = (const float*)d_in[0];
    const float* w_hh  = (const float*)d_in[1];
    const float* b_ih  = (const float*)d_in[2];
    const float* b_hh  = (const float*)d_in[3];
    const float* w_lin = (const float*)d_in[4];
    const float* b_lin = (const float*)d_in[5];
    const float* u     = (const float*)d_in[6];
    const int*   train = (const int*)d_in[8];
    float* out = (float*)d_out;

    dim3 grid(BATCH / BROWS);     // 128 blocks, independent 64-row tiles
    dim3 block(NTHREADS);         // 4 wave32 per block
    policy_lstm_scan_kernel<<<grid, block, SMEM_BYTES, stream>>>(
        w_ih, w_hh, b_ih, b_hh, w_lin, b_lin, u, train, out);
}
